// LSTMContextEncoder_45286135169709
// MI455X (gfx1250) — compile-verified, matrix operand order checked by reading, not run on hardware
//
#include <hip/hip_runtime.h>
#include <hip/hip_bf16.h>

typedef _Float16 v16h __attribute__((ext_vector_type(16)));
typedef float    v8f  __attribute__((ext_vector_type(8)));
typedef unsigned int u32x4 __attribute__((ext_vector_type(4)));
typedef int          i32x4 __attribute__((ext_vector_type(4)));
typedef int          i32x8 __attribute__((ext_vector_type(8)));

#define B_   4
#define N_   40
#define T_   64
#define NIN_ 4
#define NEMB_ 32
#define NHID_ 64
#define M_   160          // B_*N_
#define EPS_ 1e-5f

#if defined(__gfx1250__) && __has_builtin(__builtin_amdgcn_tensor_load_to_lds)
#define USE_TDM 1
#else
#define USE_TDM 0
#endif

// ---------------------------------------------------------------------------
// WMMA helpers (CDNA5, wave32). Layouts per ISA 7.12.2:
// A (16xK f16): lanes 0-15 hold row M=lane, K in {0..7} then {16..23};
//               lanes 16-31 hold row M=lane-16, K in {8..15} then {24..31}.
// B (Kx16 f16): mirrored with N = lane group.
// C/D (16x16 f32, 8 vgprs): VGPR r, lane l -> M = r + 8*(l>=16), N = l%16.
// ---------------------------------------------------------------------------
__device__ __forceinline__ v8f wmma_f16(v16h a, v16h b, v8f c) {
  return __builtin_amdgcn_wmma_f32_16x16x32_f16(false, a, false, b,
                                                (short)0, c, false, false);
}

__device__ __forceinline__ v16h load_tile_a(const _Float16* base, int stride,
                                            int row0, int kofs, int lane) {
  const int khalf = (lane < 16) ? 0 : 8;
  const int m = row0 + (lane & 15);
  const _Float16* p = base + m * stride + kofs + khalf;
  v16h a;
#pragma unroll
  for (int e = 0; e < 8; ++e) a[e] = p[e];
#pragma unroll
  for (int e = 0; e < 8; ++e) a[8 + e] = p[16 + e];
  return a;
}

__device__ __forceinline__ v16h load_tile_b_kn(const _Float16* base, int stride,
                                               int col0, int kofs, int lane) {
  const int khalf = (lane < 16) ? 0 : 8;
  const int n = col0 + (lane & 15);
  v16h b;
#pragma unroll
  for (int e = 0; e < 8; ++e) b[e] = base[(kofs + khalf + e) * stride + n];
#pragma unroll
  for (int e = 0; e < 8; ++e) b[8 + e] = base[(kofs + 16 + khalf + e) * stride + n];
  return b;
}

__device__ __forceinline__ v16h load_tile_b_nk_f32(const float* base, int stride,
                                                   int col0, int kofs, int lane) {
  const int khalf = (lane < 16) ? 0 : 8;
  const int n = col0 + (lane & 15);
  const float* p = base + n * stride + kofs + khalf;
  v16h b;
#pragma unroll
  for (int e = 0; e < 8; ++e) b[e] = (_Float16)p[e];
#pragma unroll
  for (int e = 0; e < 8; ++e) b[8 + e] = (_Float16)p[16 + e];
  return b;
}

__device__ __forceinline__ float sigmoidf_(float x) {
  return 1.0f / (1.0f + __expf(-x));
}

#if USE_TDM
// Issue a TDM linear copy: 2560 dwords (one timestep of embeddings, 10240 B)
// from global memory into LDS at byte offset lds_off (allocation-relative).
// D# group0: count=1, lds_addr, global_addr[56:0], type=2.
// D# group1: data_size=2 (4B), dim0 = tile0 = 2560, dim1 = tile1 = 1,
//            dim0_stride = 2560, workgroup_mask=0 (not in a cluster).
__device__ __forceinline__ void tdm_load_x(const _Float16* gsrc, unsigned lds_off) {
  const unsigned long long ga = (unsigned long long)gsrc;
  u32x4 g0;
  g0.x = 1u;                                   // count=1, user mode
  g0.y = lds_off;                              // lds_addr
  g0.z = (unsigned)(ga & 0xffffffffull);       // global_addr[31:0]
  g0.w = (unsigned)((ga >> 32) & 0x1ffffffull) // global_addr[56:32]
         | (2u << 30);                         // type = 2 ("image")
  i32x8 g1;
  g1[0] = 0x00020000;                          // data_size=2 (4B), mask=0
  g1[1] = (int)(2560u << 16);                  // tensor_dim0[15:0] -> bits 63:48
  g1[2] = (int)(1u << 16);                     // tensor_dim0 hi=0, tensor_dim1=1
  g1[3] = (int)(2560u << 16);                  // tile_dim0 = 2560
  g1[4] = 1;                                   // tile_dim1 = 1, tile_dim2 = 0
  g1[5] = 2560;                                // tensor_dim0_stride (low 32)
  g1[6] = 0;
  g1[7] = 0;
  i32x4 z4 = {0, 0, 0, 0};
#if __clang_major__ >= 23
  i32x8 z8 = {0, 0, 0, 0, 0, 0, 0, 0};
  __builtin_amdgcn_tensor_load_to_lds(g0, g1, z4, z4, z8, 0);
#else
  __builtin_amdgcn_tensor_load_to_lds(g0, g1, z4, z4, 0);
#endif
}
#endif

// ---------------------------------------------------------------------------
// Kernel 1: embeddings, f32 -> f16, laid out [t][m][e] for A-tile consumption.
// ---------------------------------------------------------------------------
__global__ void embed_kernel(const float* __restrict__ inputs,
                             const float* __restrict__ W_emb,
                             const float* __restrict__ b_emb,
                             _Float16* __restrict__ emb) {
  int idx = blockIdx.x * blockDim.x + threadIdx.x;  // t*5120 + m*32 + e
  if (idx >= T_ * M_ * NEMB_) return;
  const int e = idx & (NEMB_ - 1);
  const int m = (idx >> 5) % M_;
  const int t = idx / (M_ * NEMB_);
  const float* xp = inputs + (m * T_ + t) * NIN_;
  float s = b_emb[e];
#pragma unroll
  for (int i = 0; i < NIN_; ++i) s += xp[i] * W_emb[e * NIN_ + i];
  emb[idx] = (_Float16)s;
}

// ---------------------------------------------------------------------------
// Kernel 2: persistent single-workgroup LSTM. 512 threads = 16 waves.
// Wave w owns gate-column tile n0 = 16*w; its three f16 weight B-tiles stay
// in VGPRs for all 64 steps. h (f16) / c (f32) live in LDS. x_t is staged by
// the Tensor Data Mover into a double buffer (TENSORcnt + barrier), so the
// t+1 DMA overlaps step t's WMMAs. Gates staged in two 80-row LDS halves.
// Dynamic LDS layout (163840 B = 160 KiB):
//   gates: 80*256 f32 = 81920 | cbuf: 160*64 f32 = 40960
//   hbuf : 160*64 f16 = 20480 | xbuf0/1: 2 x 160*32 f16 = 2 x 10240
// ---------------------------------------------------------------------------
__global__ void lstm_kernel(const _Float16* __restrict__ emb,
                            const float* __restrict__ W_ih,
                            const float* __restrict__ W_hh,
                            const float* __restrict__ b_ih,
                            const float* __restrict__ b_hh,
                            float* __restrict__ hs) {   // [m][t][h] f32
  extern __shared__ char smem[];
  float*    gates = (float*)smem;                 // [80][256]
  float*    cbuf  = (float*)(smem + 81920);       // [160][64]
  _Float16* hbuf  = (_Float16*)(smem + 122880);   // [160][64]
  _Float16* xbuf0 = (_Float16*)(smem + 143360);   // [160][32]
  _Float16* xbuf1 = (_Float16*)(smem + 153600);   // [160][32]

  const int tid  = threadIdx.x;
  const int lane = tid & 31;
  const int w    = tid >> 5;     // 0..15
  const int n0   = w * 16;

  const v16h Bih  = load_tile_b_nk_f32(W_ih, NEMB_, n0, 0,  lane);
  const v16h Bhh0 = load_tile_b_nk_f32(W_hh, NHID_, n0, 0,  lane);
  const v16h Bhh1 = load_tile_b_nk_f32(W_hh, NHID_, n0, 32, lane);
  const int  ncol = n0 + (lane & 15);
  const float biasv = b_ih[ncol] + b_hh[ncol];

#if USE_TDM
  const unsigned xbase_off = (unsigned)__builtin_amdgcn_groupstaticsize() + 143360u;
  if (w == 0) tdm_load_x(emb, xbase_off);         // prefetch t = 0 into buf 0
#endif

  for (int idx = tid; idx < M_ * NHID_; idx += 512) {
    cbuf[idx] = 0.0f;
    hbuf[idx] = (_Float16)0.0f;
  }
  __syncthreads();

  for (int t = 0; t < T_; ++t) {
    _Float16* xb = (t & 1) ? xbuf1 : xbuf0;
#if USE_TDM
    if (w == 0) __builtin_amdgcn_s_wait_tensorcnt(0);  // buf(t) resident
    __syncthreads();
    if ((w == 0) && (t + 1 < T_))                      // overlap DMA of t+1
      tdm_load_x(emb + (t + 1) * (M_ * NEMB_),
                 xbase_off + (unsigned)(((t + 1) & 1) * (M_ * NEMB_ * 2)));
#else
    const _Float16* xe = emb + t * (M_ * NEMB_);
    for (int idx = tid; idx < (M_ * NEMB_ / 8); idx += 512)
      ((unsigned long long*)xb)[idx] = ((const unsigned long long*)xe)[idx];
    if (t + 1 < T_)
      __builtin_prefetch(emb + (t + 1) * (M_ * NEMB_), 0, 1);
    __syncthreads();
#endif

    for (int mh = 0; mh < 2; ++mh) {              // rows [mh*80, mh*80+80)
#pragma unroll
      for (int mt5 = 0; mt5 < 5; ++mt5) {
        const int mt   = mh * 5 + mt5;
        const int row0 = mt * 16;
        v8f acc = {};
        v16h ax  = load_tile_a(xb,   NEMB_, row0, 0,  lane);
        acc = wmma_f16(ax, Bih, acc);
        v16h ah0 = load_tile_a(hbuf, NHID_, row0, 0,  lane);
        acc = wmma_f16(ah0, Bhh0, acc);
        v16h ah1 = load_tile_a(hbuf, NHID_, row0, 32, lane);
        acc = wmma_f16(ah1, Bhh1, acc);
        const int mloc = mt5 * 16 + ((lane >= 16) ? 8 : 0);
#pragma unroll
        for (int r = 0; r < 8; ++r)
          gates[(mloc + r) * 256 + ncol] = acc[r] + biasv;
      }
      __syncthreads();

      for (int idx = tid; idx < 80 * NHID_; idx += 512) {
        const int ml = idx >> 6;
        const int hc = idx & 63;
        const int m  = mh * 80 + ml;
        float ig = sigmoidf_(gates[ml * 256 + hc]);
        float fg = sigmoidf_(gates[ml * 256 + 64 + hc]);
        float gg = tanhf(gates[ml * 256 + 128 + hc]);
        float og = sigmoidf_(gates[ml * 256 + 192 + hc]);
        float c  = fg * cbuf[m * NHID_ + hc] + ig * gg;
        cbuf[m * NHID_ + hc] = c;
        float h  = og * tanhf(c);
        hbuf[m * NHID_ + hc] = (_Float16)h;
        hs[m * (T_ * NHID_) + t * NHID_ + hc] = h;
      }
      __syncthreads();
    }
  }
}

// ---------------------------------------------------------------------------
// Kernel 3: per-batch context. 4 blocks x 128 threads (4 waves).
// Temporal attention collapses (row term cancels in the row-wise softmax):
// cs_last[m,h] = sum_t softmax_t(a_r[m,t]) h[m,t,h].
// Spatial: ch[b] = sum_t Adj_t(40x40) @ H_t(40x64) as one WMMA GEMM with K
// concatenated over t; per-wave 3 resident f32 accumulator tiles.
// ---------------------------------------------------------------------------
__global__ void context_kernel(const float* __restrict__ inputs,
                               const float* __restrict__ hs,
                               const float* __restrict__ W_att,
                               const float* __restrict__ b_att,
                               float* __restrict__ out) {
  __shared__ float    a_lds[N_ * T_];
  __shared__ float    wr_lds[NHID_];
  __shared__ _Float16 A_lds[48 * 64];      // padded adjacency (i x j)
  __shared__ _Float16 H_lds[64 * 64];      // padded H_t      (j x h)
  __shared__ float    loc_lds[2 * N_];

  const int b    = blockIdx.x;
  const int tid  = threadIdx.x;
  const int lane = tid & 31;
  const int w    = tid >> 5;               // 0..3
  const float* hsb = hs + b * N_ * (T_ * NHID_);

  if (tid < NHID_) wr_lds[tid] = W_att[NHID_ + tid];
  for (int idx = tid; idx < 48 * 64; idx += 128) A_lds[idx] = (_Float16)0.0f;
  for (int idx = tid; idx < 64 * 64; idx += 128) H_lds[idx] = (_Float16)0.0f;
  __syncthreads();

  for (int idx = tid; idx < N_ * T_; idx += 128) {
    const int m = idx >> 6, t = idx & 63;
    const float* hp = hsb + m * (T_ * NHID_) + t * NHID_;
    float s = b_att[0];
    for (int h2 = 0; h2 < NHID_; ++h2) s += hp[h2] * wr_lds[h2];
    a_lds[idx] = s;
  }
  __syncthreads();
  if (tid < N_) {
    float mx = -1e30f;
    for (int t = 0; t < T_; ++t) mx = fmaxf(mx, a_lds[tid * T_ + t]);
    float sum = 0.0f;
    for (int t = 0; t < T_; ++t) {
      float e = __expf(a_lds[tid * T_ + t] - mx);
      a_lds[tid * T_ + t] = e;
      sum += e;
    }
    const float inv = 1.0f / sum;
    for (int t = 0; t < T_; ++t) a_lds[tid * T_ + t] *= inv;
  }
  __syncthreads();
  for (int idx = tid; idx < N_ * NHID_; idx += 128) {
    const int m = idx >> 6, h2 = idx & 63;
    float s = 0.0f;
    for (int t = 0; t < T_; ++t)
      s += a_lds[m * T_ + t] * hsb[m * (T_ * NHID_) + t * NHID_ + h2];
    out[(b * N_ + m) * 128 + h2] = tanhf(s);
  }

  v8f acc0 = {}, acc1 = {}, acc2 = {};
  for (int t = 0; t < T_; ++t) {
    __syncthreads();
    if (tid < 2 * N_)
      loc_lds[tid] = inputs[((b * N_ + (tid >> 1)) * T_ + t) * NIN_ + (tid & 1)];
    for (int idx = tid; idx < N_ * NHID_; idx += 128) {
      const int j = idx >> 6, h2 = idx & 63;
      H_lds[j * 64 + h2] = (_Float16)hsb[j * (T_ * NHID_) + t * NHID_ + h2];
    }
    if (t + 1 < T_)
      __builtin_prefetch(hsb + (t + 1) * NHID_, 0, 1);  // global_prefetch_b8
    __syncthreads();
    for (int idx = tid; idx < N_ * N_; idx += 128) {
      const int i = idx / N_, j = idx - i * N_;
      const float dx = loc_lds[2 * i] - loc_lds[2 * j];
      const float dy = loc_lds[2 * i + 1] - loc_lds[2 * j + 1];
      const float d  = sqrtf(dx * dx + dy * dy);
      A_lds[i * 64 + j] = (_Float16)((i == j) ? 0.0f : 1.0f / (d + EPS_));
    }
    __syncthreads();
    const v16h b0 = load_tile_b_kn(H_lds, 64, w * 16, 0,  lane);
    const v16h b1 = load_tile_b_kn(H_lds, 64, w * 16, 32, lane);
    v16h a;
    a = load_tile_a(A_lds, 64, 0,  0,  lane); acc0 = wmma_f16(a, b0, acc0);
    a = load_tile_a(A_lds, 64, 0,  32, lane); acc0 = wmma_f16(a, b1, acc0);
    a = load_tile_a(A_lds, 64, 16, 0,  lane); acc1 = wmma_f16(a, b0, acc1);
    a = load_tile_a(A_lds, 64, 16, 32, lane); acc1 = wmma_f16(a, b1, acc1);
    a = load_tile_a(A_lds, 64, 32, 0,  lane); acc2 = wmma_f16(a, b0, acc2);
    a = load_tile_a(A_lds, 64, 32, 32, lane); acc2 = wmma_f16(a, b1, acc2);
  }
  __syncthreads();

  const int hc   = w * 16 + (lane & 15);
  const int roff = (lane >= 16) ? 8 : 0;
#pragma unroll
  for (int r = 0; r < 8; ++r) {
    const int i0 = r + roff;
    out[(b * N_ + i0) * 128 + 64 + hc]      = tanhf(acc0[r]);
    out[(b * N_ + 16 + i0) * 128 + 64 + hc] = tanhf(acc1[r]);
    if (32 + i0 < N_)
      out[(b * N_ + 32 + i0) * 128 + 64 + hc] = tanhf(acc2[r]);
  }
}

// ---------------------------------------------------------------------------
extern "C" void kernel_launch(void* const* d_in, const int* in_sizes, int n_in,
                              void* d_out, int out_size, void* d_ws, size_t ws_size,
                              hipStream_t stream) {
  const float* inputs = (const float*)d_in[0];
  const float* W_emb  = (const float*)d_in[5];
  const float* b_emb  = (const float*)d_in[6];
  const float* W_ih   = (const float*)d_in[7];
  const float* W_hh   = (const float*)d_in[8];
  const float* b_ih   = (const float*)d_in[9];
  const float* b_hh   = (const float*)d_in[10];
  const float* W_att  = (const float*)d_in[11];
  const float* b_att  = (const float*)d_in[12];
  float* out = (float*)d_out;

  _Float16* emb = (_Float16*)d_ws;                              // 655360 B
  float* hs = (float*)((char*)d_ws + (size_t)T_ * M_ * NEMB_ * sizeof(_Float16));

  const int n_emb = T_ * M_ * NEMB_;
  embed_kernel<<<(n_emb + 255) / 256, 256, 0, stream>>>(inputs, W_emb, b_emb, emb);

  const size_t lstm_lds = 163840;  // 160 KiB, see layout above
  lstm_kernel<<<1, 512, lstm_lds, stream>>>(emb, W_ih, W_hh, b_ih, b_hh, hs);

  context_kernel<<<B_, 128, 0, stream>>>(inputs, hs, W_att, b_att, out);
}
